// StrengthNet_61246233641703
// MI455X (gfx1250) — matrix operand order repaired and checked by reading, not compiled, not run on hardware
//
#include <hip/hip_runtime.h>
#include <hip/hip_bf16.h>
#include <stdint.h>

// ---------------------------------------------------------------------------
// Set Transformer (ISAB x2 + PMA pool) for MI455X / gfx1250, wave32 + WMMA.
// - All GEMMs: v_wmma_f32_16x16x32_f16 (f16 operands, fp32 accumulate).
// - Activations kept as resident f16 in HBM (fp32 masters only where LN needs
//   them) -> halves HBM traffic vs fp32 (23.3 TB/s roofline).
// - Global->LDS staging via GLOBAL_LOAD_ASYNC_TO_LDS_B128 (ASYNCcnt) and the
//   Tensor Data Mover (tensor_load_to_lds, TENSORcnt) for contiguous V tiles.
// - Softmax associativity: a @ (h Wv^T) == (a @ h) @ Wv^T for Lq << Lk blocks.
// - 1/sqrt(QD) folded into the K-projection GEMM epilogue.
// ---------------------------------------------------------------------------

typedef __attribute__((ext_vector_type(16))) _Float16     v16h;
typedef __attribute__((ext_vector_type(8)))  float        v8f;
typedef __attribute__((ext_vector_type(4)))  unsigned int u32x4;
typedef __attribute__((ext_vector_type(4)))  int          i32x4;
typedef __attribute__((ext_vector_type(8)))  int          i32x8;

#define DEV static __device__ __forceinline__

DEV v8f v8f_zero() {
  v8f v;
#pragma unroll
  for (int i = 0; i < 8; ++i) v[i] = 0.0f;
  return v;
}

DEV v8f wmma16(v16h a, v16h b, v8f c) {
  return __builtin_amdgcn_wmma_f32_16x16x32_f16(
      false, a, false, b, (short)0, c, false, false);
}

// LDS byte offset of a __shared__ pointer (flat LDS addresses carry the
// workgroup-relative offset in the low 32 bits).
DEV unsigned lds_off(const void* p) { return (unsigned)(uintptr_t)p; }

// Issue one 16-byte async global->LDS copy (per-lane), tracked by ASYNCcnt.
DEV void async_copy_b128(unsigned lds_byte_off, const void* gaddr) {
  asm volatile("global_load_async_to_lds_b128 %0, %1, off"
               :: "v"(lds_byte_off), "v"(gaddr) : "memory");
}
DEV void wait_asynccnt0() {
  asm volatile("s_wait_asynccnt 0x0" ::: "memory");
}

// A-fragment (16x32 f16): lane l holds row M = mbase+(l&15);
// halfs 0..15 -> K=(l>>4)*16+0..15
DEV v16h load_a_frag(const _Float16* lds, int stride, int mbase, int kbase) {
  int l  = threadIdx.x & 31;
  int m  = mbase + (l & 15);
  int kb = (l >> 4) * 16;
  const _Float16* p = lds + (size_t)m * stride + kbase;
  v16h a;
#pragma unroll
  for (int j = 0; j < 8; ++j) {
    a[j]     = p[kb + j];
    a[8 + j] = p[8 + kb + j];
  }
  return a;
}

// B-fragment from N x K ("transposed") storage: B[k][n] = lds[n][k]
DEV v16h load_b_frag_T(const _Float16* lds, int stride, int nbase, int kbase) {
  int l   = threadIdx.x & 31;
  int n   = nbase + (l & 15);
  int kb2 = (l >> 4) * 16;
  const _Float16* p = lds + (size_t)n * stride + kbase + kb2;
  v16h b;
#pragma unroll
  for (int j = 0; j < 16; ++j) b[j] = p[j];
  return b;
}

// B-fragment from K x N (natural) storage: B[k][n] = lds[k][n]
DEV v16h load_b_frag_N(const _Float16* lds, int stride, int nbase, int kbase) {
  int l   = threadIdx.x & 31;
  int n   = nbase + (l & 15);
  int kb2 = (l >> 4) * 16;
  const _Float16* p = lds + (size_t)(kbase + kb2) * stride + n;
  v16h b;
#pragma unroll
  for (int j = 0; j < 16; ++j) b[j] = p[(size_t)j * stride];
  return b;
}

// ---------------------------------------------------------------------------
// GEMM: C(NxE) = oscale * (A(NxK) @ W(ExK)^T) + bias.  A is f16, W/bias fp32.
// Outputs optionally fp32 and/or f16.  N%64==0, K%32==0, E%64==0.
// Block = 128 threads (4 waves), 64x64 tile. A tile staged via async-to-LDS.
// ---------------------------------------------------------------------------
__global__ void gemm_f16a_kernel(const _Float16* __restrict__ A,
                                 const float* __restrict__ W,
                                 const float* __restrict__ bias, float oscale,
                                 float* __restrict__ C32,
                                 _Float16* __restrict__ C16,
                                 int N, int K, int E) {
  __shared__ _Float16 la[64][40];   // 64x32 A tile (+pad, 80B row = 16B-aligned)
  __shared__ _Float16 lw[64][40];   // 64x32 W tile (+pad)
  const int wave  = threadIdx.x >> 5;
  const int lane  = threadIdx.x & 31;
  const int nBase = blockIdx.x * 64;
  const int eBase = blockIdx.y * 64;

  v8f acc[4];
#pragma unroll
  for (int t = 0; t < 4; ++t) acc[t] = v8f_zero();

  for (int k0 = 0; k0 < K; k0 += 32) {
    // A tile: 64 rows x 32 halfs = 256 x 16B granules, async to LDS
    for (int i = threadIdx.x; i < 256; i += 128) {
      int r = i >> 2, g = i & 3;
      async_copy_b128(lds_off(&la[r][g * 8]),
                      (const void*)(A + (size_t)(nBase + r) * K + k0 + g * 8));
    }
    // W tile: fp32 -> f16 conversion while staging
    for (int i = threadIdx.x; i < 64 * 32; i += 128) {
      int r = i >> 5, c = i & 31;
      lw[r][c] = (_Float16)W[(size_t)(eBase + r) * K + k0 + c];
    }
    if (k0 + 32 < K) {   // prefetch next K-step (global_prefetch_b8)
      __builtin_prefetch((const void*)(A + (size_t)(nBase + (threadIdx.x & 63)) * K + k0 + 32), 0, 3);
      __builtin_prefetch((const void*)(W + (size_t)(eBase + (threadIdx.x & 63)) * K + k0 + 32), 0, 3);
    }
    wait_asynccnt0();
    __syncthreads();
    v16h af = load_a_frag(&la[0][0], 40, wave * 16, 0);
#pragma unroll
    for (int t = 0; t < 4; ++t) {
      v16h bf = load_b_frag_T(&lw[0][0], 40, t * 16, 0);
      acc[t] = wmma16(af, bf, acc[t]);
    }
    __syncthreads();
  }
#pragma unroll
  for (int t = 0; t < 4; ++t) {
    int col = eBase + t * 16 + (lane & 15);
    float bv = bias ? bias[col] : 0.0f;
#pragma unroll
    for (int r = 0; r < 8; ++r) {
      int row = nBase + wave * 16 + ((lane >> 4) * 8) + r;
      size_t idx = (size_t)row * E + col;
      float v = acc[t][r] * oscale + bv;
      if (C32) C32[idx] = v;
      if (C16) C16[idx] = (_Float16)v;
    }
  }
}

// ---------------------------------------------------------------------------
// Batched attention core. Per block: one (b, 32-q-row tile).
// logits = qq @ K^T (scale pre-folded into K), LDS softmax, U = a @ V.
// QQ/K/V are f16 in global. K chunk staged via async-to-LDS; V chunk (16KB,
// contiguous) staged via the Tensor Data Mover when available.
// Dynamic LDS: logits(32*Lk f32) | qq(32x64) | kc(32x64) | a(32*Lk) | vc(32x256)
// ---------------------------------------------------------------------------
__global__ void attn_kernel(const _Float16* __restrict__ QQ, int qq_bstride,
                            const _Float16* __restrict__ Km,
                            const _Float16* __restrict__ Vm,
                            float* __restrict__ U32,
                            _Float16* __restrict__ U16,
                            int Lq, int Lk, int nQT) {
  extern __shared__ char smem[];
  float*    logits = (float*)smem;
  _Float16* qqh = (_Float16*)(smem + (size_t)32 * Lk * 4);
  _Float16* kc  = qqh + 32 * 64;
  _Float16* ah  = kc + 32 * 64;
  _Float16* vc  = ah + (size_t)32 * Lk;

  const int b     = blockIdx.x / nQT;
  const int qBase = (blockIdx.x % nQT) * 32;
  const int tid   = threadIdx.x;
  const int wave  = tid >> 5;
  const int lane  = tid & 31;

  // ---- q tile (f16 copy, zero-pad rows >= Lq) ----
  const _Float16* QQb = QQ + (size_t)b * qq_bstride;
  for (int i = tid; i < 32 * 64; i += 256) {
    int r = i >> 6, c = i & 63;
    int gr = qBase + r;
    qqh[i] = (gr < Lq) ? QQb[(size_t)gr * 64 + c] : (_Float16)0.0f;
  }
  __syncthreads();

  // ---- phase 1: logits = qq @ K^T ----
  const _Float16* Kb = Km + (size_t)b * Lk * 64;
  for (int jc = 0; jc < Lk; jc += 32) {
    {  // K chunk: 32 rows x 64 halfs = 256 x 16B granules, async to LDS
      int r = tid >> 3, g = tid & 7;
      async_copy_b128(lds_off(kc + r * 64 + g * 8),
                      (const void*)(Kb + (size_t)(jc + r) * 64 + g * 8));
    }
    wait_asynccnt0();
    __syncthreads();
    if (wave < 4) {
      int rt = wave >> 1, ct = wave & 1;
      v8f acc = v8f_zero();
#pragma unroll
      for (int ks = 0; ks < 2; ++ks) {
        v16h af = load_a_frag(qqh, 64, rt * 16, ks * 32);
        v16h bf = load_b_frag_T(kc, 64, ct * 16, ks * 32);
        acc = wmma16(af, bf, acc);
      }
#pragma unroll
      for (int r = 0; r < 8; ++r) {
        int mr = rt * 16 + ((lane >> 4) * 8) + r;
        int nn = jc + ct * 16 + (lane & 15);
        logits[(size_t)mr * Lk + nn] = acc[r];
      }
    }
    __syncthreads();
  }

  // ---- phase 2: row softmax (each wave owns 4 rows) ----
  for (int rr = 0; rr < 4; ++rr) {
    int row = wave * 4 + rr;
    float* lr = logits + (size_t)row * Lk;
    float mx = -1e30f;
    for (int j = lane; j < Lk; j += 32) mx = fmaxf(mx, lr[j]);
#pragma unroll
    for (int o = 16; o; o >>= 1) mx = fmaxf(mx, __shfl_xor(mx, o, 32));
    float sum = 0.0f;
    for (int j = lane; j < Lk; j += 32) {
      float e = __expf(lr[j] - mx);
      lr[j] = e;
      sum += e;
    }
#pragma unroll
    for (int o = 16; o; o >>= 1) sum += __shfl_xor(sum, o, 32);
    float inv = 1.0f / sum;
    for (int j = lane; j < Lk; j += 32)
      ah[(size_t)row * Lk + j] = (_Float16)(lr[j] * inv);
  }
  __syncthreads();

  // ---- phase 3: U = a @ V  (V is Lk x 256 f16) ----
  const _Float16* Vb = Vm + (size_t)b * Lk * 256;
  const int rt = wave >> 2;
  const int cg = wave & 3;
  v8f acc[4];
#pragma unroll
  for (int t = 0; t < 4; ++t) acc[t] = v8f_zero();
  for (int jc = 0; jc < Lk; jc += 32) {
    // V chunk = 32x256 f16 = 16KB, contiguous in global: 1-D TDM tile.
#if __has_builtin(__builtin_amdgcn_tensor_load_to_lds)
    if (wave == 0) {
      unsigned long long ga =
          (unsigned long long)(uintptr_t)(const void*)(Vb + (size_t)jc * 256);
      u32x4 g0;
      g0.x = 1u;                                   // count=1 (valid descriptor)
      g0.y = lds_off(vc);                          // lds_addr (bytes)
      g0.z = (unsigned)ga;                         // global_addr[31:0]
      g0.w = ((unsigned)(ga >> 32) & 0x01FFFFFFu)  // global_addr[56:32]
             | (2u << 30);                         // type=2 ("image")
      i32x8 g1;
      g1[0] = 0x00010000;                          // data_size=1 (2 bytes)
      g1[1] = (int)(8192u << 16);                  // tensor_dim0[15:0] -> [31:16]
      g1[2] = (int)(8192u >> 16);                  // tensor_dim0[31:16]
      g1[3] = (int)(8192u << 16);                  // tile_dim0=8192 -> [31:16]
      g1[4] = 0;                                   // tile_dim1/2 unused (1-D)
      g1[5] = 8192;                                // tensor_dim0_stride[31:0]
      g1[6] = 0;
      g1[7] = 0;
      i32x4 z4; z4[0] = 0; z4[1] = 0; z4[2] = 0; z4[3] = 0;
#if defined(__clang_major__) && __clang_major__ >= 23
      i32x8 z8;
#pragma unroll
      for (int q = 0; q < 8; ++q) z8[q] = 0;
      __builtin_amdgcn_tensor_load_to_lds(g0, g1, z4, z4, z8, 0);
#else
      __builtin_amdgcn_tensor_load_to_lds(g0, g1, z4, z4, 0);
#endif
      __builtin_amdgcn_s_wait_tensorcnt(0);
    }
#else
    for (int i = tid; i < 1024; i += 256) {        // async-to-LDS fallback
      int r = i >> 5, g = i & 31;
      async_copy_b128(lds_off(vc + r * 256 + g * 8),
                      (const void*)(Vb + (size_t)(jc + r) * 256 + g * 8));
    }
    wait_asynccnt0();
#endif
    __syncthreads();
    v16h af = load_a_frag(ah, Lk, rt * 16, jc);
#pragma unroll
    for (int t = 0; t < 4; ++t) {
      v16h bf = load_b_frag_N(vc, 256, cg * 64 + t * 16, 0);
      acc[t] = wmma16(af, bf, acc[t]);
    }
    __syncthreads();
  }
#pragma unroll
  for (int t = 0; t < 4; ++t) {
#pragma unroll
    for (int r = 0; r < 8; ++r) {
      int mr  = rt * 16 + ((lane >> 4) * 8) + r;
      int row = qBase + mr;
      if (row < Lq) {
        int col = cg * 64 + t * 16 + (lane & 15);
        size_t idx = ((size_t)b * Lq + row) * 256 + col;
        float v = acc[t][r];
        if (U32) U32[idx] = v;
        if (U16) U16[idx] = (_Float16)v;
      }
    }
  }
}

// ---------------------------------------------------------------------------
// out = LayerNorm(base + f(res)) * g + b ; f = relu or identity. Row width 256.
// One wave per row. base_mod!=0 => base broadcast (base has base_mod rows).
// Writes fp32 and/or f16 outputs.
// ---------------------------------------------------------------------------
__global__ void addln_kernel(const float* __restrict__ base, int base_mod,
                             const float* __restrict__ res, int relu,
                             const float* __restrict__ g,
                             const float* __restrict__ bt,
                             float* __restrict__ out32,
                             _Float16* __restrict__ out16, int nrows) {
  const int wave = threadIdx.x >> 5, lane = threadIdx.x & 31;
  const int row = blockIdx.x * 8 + wave;
  if (row >= nrows) return;
  const int brow = base_mod ? (row % base_mod) : row;
  const float* bp = base + (size_t)brow * 256;
  const float* rp = res + (size_t)row * 256;
  float v[8];
  float s = 0.0f;
#pragma unroll
  for (int k = 0; k < 8; ++k) {
    int c = lane + 32 * k;
    float r = rp[c];
    if (relu) r = fmaxf(r, 0.0f);
    v[k] = bp[c] + r;
    s += v[k];
  }
#pragma unroll
  for (int o = 16; o; o >>= 1) s += __shfl_xor(s, o, 32);
  const float mean = s * (1.0f / 256.0f);
  float q = 0.0f;
#pragma unroll
  for (int k = 0; k < 8; ++k) {
    float d = v[k] - mean;
    q += d * d;
  }
#pragma unroll
  for (int o = 16; o; o >>= 1) q += __shfl_xor(q, o, 32);
  const float inv = rsqrtf(q * (1.0f / 256.0f) + 1e-5f);
#pragma unroll
  for (int k = 0; k < 8; ++k) {
    int c = lane + 32 * k;
    float y = (v[k] - mean) * inv * g[c] + bt[c];
    if (out32) out32[(size_t)row * 256 + c] = y;
    if (out16) out16[(size_t)row * 256 + c] = (_Float16)y;
  }
}

// Tiny shared-query projection: out16(nq x 64) = q(nq x 256) @ wq(64 x 256)^T
__global__ void small_proj_kernel(const float* __restrict__ q,
                                  const float* __restrict__ wq,
                                  _Float16* __restrict__ out, int nq) {
  int t = blockIdx.x * blockDim.x + threadIdx.x;
  int r = t >> 6, e = t & 63;
  if (r >= nq) return;
  float s = 0.0f;
  for (int k = 0; k < 256; ++k) s += q[(size_t)r * 256 + k] * wq[(size_t)e * 256 + k];
  out[(size_t)r * 64 + e] = (_Float16)s;
}

// f32 -> f16 elementwise (for network input x)
__global__ void cvt_f16_kernel(const float* __restrict__ in,
                               _Float16* __restrict__ out, int n) {
  for (int i = blockIdx.x * blockDim.x + threadIdx.x; i < n;
       i += gridDim.x * blockDim.x)
    out[i] = (_Float16)in[i];
}

// out[b] = dot(pooled[b, :256], w[:256])
__global__ void final_dot_kernel(const float* __restrict__ pooled,
                                 const float* __restrict__ w,
                                 float* __restrict__ out, int Bn) {
  const int wave = threadIdx.x >> 5, lane = threadIdx.x & 31;
  const int b = blockIdx.x * 8 + wave;
  if (b >= Bn) return;
  float s = 0.0f;
  for (int k = lane; k < 256; k += 32) s += pooled[(size_t)b * 256 + k] * w[k];
#pragma unroll
  for (int o = 16; o; o >>= 1) s += __shfl_xor(s, o, 32);
  if (lane == 0) out[b] = s;
}

// ---------------------------------------------------------------------------
// Host orchestration
// ---------------------------------------------------------------------------
struct ABP { const float *b0,*b1,*fb,*fw,*g0,*g1,*wk,*wq,*wv; };

static ABP read_ab(void* const* d_in, int& i) {
  ABP p;
  p.b0 = (const float*)d_in[i++]; p.b1 = (const float*)d_in[i++];
  p.fb = (const float*)d_in[i++]; p.fw = (const float*)d_in[i++];
  p.g0 = (const float*)d_in[i++]; p.g1 = (const float*)d_in[i++];
  p.wk = (const float*)d_in[i++]; p.wq = (const float*)d_in[i++];
  p.wv = (const float*)d_in[i++];
  return p;
}

extern "C" void kernel_launch(void* const* d_in, const int* in_sizes, int n_in,
                              void* d_out, int out_size, void* d_ws, size_t ws_size,
                              hipStream_t stream) {
  (void)in_sizes; (void)out_size; (void)ws_size;
  constexpr int Bn = 256, L = 512, D = 256, QD = 64, M = 32;
  if (n_in < 51) return;

  // Input order: x first, then params flattened in JAX pytree (sorted-key)
  // order: emb, isab[0]{ab0(9),ab1(9),i}, isab[1]{...}, out, pool(9), seed.
  int ii = 0;
  const float* x   = (const float*)d_in[ii++];
  const float* emb = (const float*)d_in[ii++];
  ABP ab[2][2]; const float* ivec[2];
  for (int l = 0; l < 2; ++l) {
    ab[l][0] = read_ab(d_in, ii);
    ab[l][1] = read_ab(d_in, ii);
    ivec[l]  = (const float*)d_in[ii++];
  }
  const float* outw = (const float*)d_in[ii++];
  ABP pool = read_ab(d_in, ii);
  const float* seed = (const float*)d_in[ii++];

  // Workspace carve (256B aligned)
  char* cur = (char*)d_ws;
  auto alloc = [&](size_t bytes) {
    void* p = cur; cur += (bytes + 255) & ~(size_t)255; return p;
  };
  const size_t BLD = (size_t)Bn * L * D, BMD = (size_t)Bn * M * D;
  float* H0 = (float*)alloc(BLD * 4);        // h fp32 master
  float* H1 = (float*)alloc(BLD * 4);        // HO / hres (wide blocks)
  float* T1 = (float*)alloc(BMD * 4);        // HO ab0
  float* T2 = (float*)alloc(BMD * 4);        // h1 ab0 fp32
  float* T3 = (float*)alloc(BMD * 4);        // hres ab0
  float* P1 = (float*)alloc((size_t)Bn * D * 4);
  float* P2 = (float*)alloc((size_t)Bn * D * 4);
  float* P3 = (float*)alloc((size_t)Bn * D * 4);
  _Float16* Hh  = (_Float16*)alloc(BLD * 2);               // h f16 mirror
  _Float16* Xh  = (_Float16*)alloc((size_t)Bn * L * 32 * 2);
  _Float16* S0h = (_Float16*)alloc((size_t)Bn * L * QD * 2); // K (scaled)
  _Float16* S1h = (_Float16*)alloc((size_t)Bn * L * QD * 2); // QQ ab1
  _Float16* T0h = (_Float16*)alloc(BMD * 2); // u ab0
  _Float16* T2h = (_Float16*)alloc(BMD * 2); // h1 ab0 f16
  _Float16* T4h = (_Float16*)alloc(BMD * 2); // hi f16
  _Float16* T5h = (_Float16*)alloc(BMD * 2); // V ab1 f16
  _Float16* P0h = (_Float16*)alloc((size_t)Bn * D * 2);
  _Float16* P2h = (_Float16*)alloc((size_t)Bn * D * 2);
  _Float16* qqsh = (_Float16*)alloc(32 * 64 * 2);

  const float kscale = 0.125f;   // 1/sqrt(QD), folded into K projection

  auto gemm = [&](const _Float16* A, const float* W, const float* bias,
                  float oscale, float* C32, _Float16* C16, int N, int K, int E) {
    hipLaunchKernelGGL(gemm_f16a_kernel, dim3(N / 64, E / 64), dim3(128), 0,
                       stream, A, W, bias, oscale, C32, C16, N, K, E);
  };
  auto attn = [&](const _Float16* QQ, int qqbs, const _Float16* Kp,
                  const _Float16* Vp, float* U32, _Float16* U16,
                  int Lq, int Lk) {
    int nQT = (Lq + 31) / 32;
    size_t lds = (size_t)32 * Lk * 4 + 4096 + 4096 + (size_t)32 * Lk * 2 + 16384;
    (void)hipFuncSetAttribute((const void*)attn_kernel,
                              hipFuncAttributeMaxDynamicSharedMemorySize,
                              (int)131072);
    hipLaunchKernelGGL(attn_kernel, dim3(Bn * nQT), dim3(256), lds, stream,
                       QQ, qqbs, Kp, Vp, U32, U16, Lq, Lk, nQT);
  };
  auto addln = [&](const float* base, int bmod, const float* res, int relu,
                   const float* g, const float* bt, float* o32, _Float16* o16,
                   int rows) {
    hipLaunchKernelGGL(addln_kernel, dim3(rows / 8), dim3(256), 0, stream,
                       base, bmod, res, relu, g, bt, o32, o16, rows);
  };

  // 0) x -> f16
  hipLaunchKernelGGL(cvt_f16_kernel, dim3(4096), dim3(256), 0, stream,
                     x, Xh, Bn * L * 32);
  // 1) embedding: h = x @ emb^T (fp32 master + f16 mirror)
  gemm(Xh, emb, nullptr, 1.0f, H0, Hh, Bn * L, 32, D);

  // 2) ISAB layers
  for (int l = 0; l < 2; ++l) {
    const ABP& a0 = ab[l][0];
    const ABP& a1 = ab[l][1];
    // --- ab0: hi = attn_block(a0, ind, h) ; Lq=32, Lk=512 ---
    hipLaunchKernelGGL(small_proj_kernel, dim3(8), dim3(256), 0, stream,
                       ivec[l], a0.wq, qqsh, 32);
    gemm(Hh, a0.wk, nullptr, kscale, nullptr, S0h, Bn * L, D, QD); // K*scale
    attn(qqsh, 0, S0h, Hh, nullptr, T0h, M, L);                    // u = a @ h
    gemm(T0h, a0.wv, nullptr, 1.0f, T1, nullptr, Bn * M, D, D);    // HO
    addln(ivec[l], M, T1, 0, a0.g0, a0.b0, T2, T2h, Bn * M);       // h1
    gemm(T2h, a0.fw, a0.fb, 1.0f, T3, nullptr, Bn * M, D, D);      // hres
    addln(T2, 0, T3, 1, a0.g1, a0.b1, nullptr, T4h, Bn * M);       // hi (f16)
    // --- ab1: h = attn_block(a1, h, hi) ; Lq=512, Lk=32 ---
    gemm(Hh, a1.wq, nullptr, 1.0f, nullptr, S1h, Bn * L, D, QD);   // QQ
    gemm(T4h, a1.wk, nullptr, kscale, nullptr, S0h, Bn * M, D, QD);// K*scale
    gemm(T4h, a1.wv, nullptr, 1.0f, nullptr, T5h, Bn * M, D, D);   // V
    attn(S1h, L * QD, S0h, T5h, H1, nullptr, L, M);                // HO
    addln(H0, 0, H1, 0, a1.g0, a1.b0, H0, Hh, Bn * L);             // h1
    gemm(Hh, a1.fw, a1.fb, 1.0f, H1, nullptr, Bn * L, D, D);       // hres
    addln(H0, 0, H1, 1, a1.g1, a1.b1, H0, Hh, Bn * L);             // new h
  }

  // 3) PMA pool: Lq=1, Lk=512  (u = a @ h, then @ wv^T)
  hipLaunchKernelGGL(small_proj_kernel, dim3(1), dim3(64), 0, stream,
                     seed, pool.wq, qqsh, 1);
  gemm(Hh, pool.wk, nullptr, kscale, nullptr, S0h, Bn * L, D, QD);
  attn(qqsh, 0, S0h, Hh, nullptr, P0h, 1, L);
  gemm(P0h, pool.wv, nullptr, 1.0f, P1, nullptr, Bn, D, D);        // HO
  addln(seed, 1, P1, 0, pool.g0, pool.b0, P2, P2h, Bn);            // h1
  gemm(P2h, pool.fw, pool.fb, 1.0f, P3, nullptr, Bn, D, D);        // hres
  addln(P2, 0, P3, 1, pool.g1, pool.b1, P1, nullptr, Bn);          // pooled
  hipLaunchKernelGGL(final_dot_kernel, dim3(Bn / 8), dim3(256), 0, stream,
                     P1, outw, (float*)d_out, Bn);
}
